// Conv4Pim_group_split_v2_10445360464135
// MI455X (gfx1250) — matrix operand order repaired and checked by reading, not compiled, not run on hardware
//
#include <hip/hip_runtime.h>
#include <hip/hip_bf16.h>
#include <stddef.h>

typedef __bf16 bf16_t;
typedef unsigned short u16_t;
typedef __attribute__((ext_vector_type(16))) __bf16 v16bf;
typedef __attribute__((ext_vector_type(8)))  float  v8f;
typedef unsigned int u32x4 __attribute__((ext_vector_type(4)));
typedef int i32x8 __attribute__((ext_vector_type(8)));
typedef int i32x4 __attribute__((ext_vector_type(4)));

#if defined(__has_builtin)
#  if __has_builtin(__builtin_amdgcn_tensor_load_to_lds) && \
      __has_builtin(__builtin_amdgcn_s_wait_tensorcnt)
#    define USE_TDM 1
#  endif
#endif
#ifndef USE_TDM
#  define USE_TDM 0
#endif

namespace {
constexpr int kB = 32, kIC = 256, kH = 32, kW = 32, kHP = 34, kWP = 34;
constexpr int kOC = 256, kG = 4;
constexpr int kKT = 2304;            // 9 kernel positions * 256 ic (pos-major)
constexpr int kCh = 1024;
constexpr int MTILE = 64, NTILE = 32, KCHUNK = 64;
constexpr int NCHUNK = kKT / KCHUNK; // 36
constexpr int THREADS = 256;         // 8 wave32 = 2 branches x 4 groups
constexpr int AROW  = 72;            // A-tile LDS row stride in u16 (144 B)
constexpr int WROWB = 144;           // W-tile LDS row stride bytes (128 + 16 pad)
constexpr size_t WQ_ELEMS = 2ull * kCh * kKT;             // packed weights
constexpr size_t XP_ELEMS = (size_t)kB * kIC * kHP * kWP; // padded bf16 x
}

union ABFrag { v16bf v; uint4 q[2]; };
union Pack8  { uint4 q; u16_t u[8]; };

// ---------------------------------------------------------------------------
// Stage 1: relu + LSQ round/clip + 2-bit slice -> packed bf16 weights.
// Wq[branch][c][k], k = (r*3+s)*256 + ic. Values {0,1,2,3} exact in bf16.
// ---------------------------------------------------------------------------
__global__ __launch_bounds__(256)
void quant_weights_kernel(const float* __restrict__ w,
                          const float* __restrict__ s_w_p,
                          const float* __restrict__ s_w_n,
                          bf16_t* __restrict__ wq) {
  long long idx = (long long)blockIdx.x * blockDim.x + threadIdx.x;
  const long long total = 2LL * kCh * kKT;
  if (idx >= total) return;
  int k  = (int)(idx % kKT);
  int c  = (int)((idx / kKT) % kCh);
  int br = (int)(idx / ((long long)kKT * kCh));
  int kpos = k >> 8, ic = k & 255;
  int r = kpos / 3, s = kpos - 3 * r;
  float wv = w[(((size_t)c * kIC + ic) * 3 + r) * 3 + s];
  float sw = br ? s_w_n[0] : s_w_p[0];
  float xp = br ? fmaxf(-wv, 0.f) : fmaxf(wv, 0.f);
  float wi = rintf(fminf(fmaxf(xp / sw, 0.f), 15.f));
  float sl = wi - 4.f * floorf(wi * 0.25f);
  wq[idx] = (bf16_t)sl;
}

// ---------------------------------------------------------------------------
// Stage 2: build zero-padded bf16 activation image [32][256][34][34].
// Removes all halo bounds checks and f32->bf16 converts from the GEMM loop.
// ---------------------------------------------------------------------------
__global__ __launch_bounds__(256)
void pad_cvt_kernel(const float* __restrict__ x, bf16_t* __restrict__ xpad) {
  long long idx = (long long)blockIdx.x * blockDim.x + threadIdx.x;
  if (idx >= (long long)XP_ELEMS) return;
  int pix = (int)(idx % (kHP * kWP));
  int c   = (int)(idx / (kHP * kWP));    // b*256 + ic
  int y = pix / kWP - 1, xx = pix % kWP - 1;
  float v = 0.f;
  if ((unsigned)y < (unsigned)kH && (unsigned)xx < (unsigned)kW)
    v = x[((size_t)c * kH + y) * kW + xx];
  xpad[idx] = (bf16_t)v;
}

// ---------------------------------------------------------------------------
// Stage 3: implicit-GEMM conv on WMMA. Block = 64(M) x 32(OC) tile; wave w of
// 8 owns (branch = w/4, group = w%4). A-tile (im2col bf16) double-buffered in
// LDS; per-wave weight tiles double-buffered via the Tensor Data Mover.
// ---------------------------------------------------------------------------
__global__ __launch_bounds__(256)
void conv_wmma_kernel(const bf16_t* __restrict__ xpad,
                      const bf16_t* __restrict__ wq,
                      const float* __restrict__ s_w_p,
                      const float* __restrict__ s_w_n,
                      const float* __restrict__ s_ps_p,
                      const float* __restrict__ s_ps_n,
                      float* __restrict__ out) {
  __shared__ __align__(16) u16_t smA[2][MTILE][AROW];           // 18 KB
#if USE_TDM
  __shared__ __align__(16) unsigned char smW[2][8][32 * WROWB]; // 72 KB
#endif
  __shared__ float smO[MTILE][NTILE];                           // 8 KB

  const int tid  = threadIdx.x;
  const int lane = tid & 31;
  const int wave = tid >> 5;
  const int br = wave >> 2;
  const int g  = wave & 3;
  const int mtile = blockIdx.x;     // 512
  const int ntile = blockIdx.y;     // 8

  const float sw      = br ? s_w_n[0] : s_w_p[0];
  const float sps     = br ? s_ps_n[g] : s_ps_p[g];
  const float inv_sps = 1.f / sps;
  const float sgn     = br ? -1.f : 1.f;

  for (int i = tid; i < MTILE * NTILE; i += THREADS)
    (&smO[0][0])[i] = 0.f;

  // A-stager: thread moves 2 k-rows x 8 pixels (16B contiguous global chunks)
  const int kk2  = tid >> 3;        // 0..31  -> kk = kk2*2 + jj
  const int mblk = tid & 7;         // m = mblk*8 + i (one h-row, consecutive w)
  const int mg0 = mtile * MTILE + mblk * 8;
  const int bb = mg0 >> 10;
  const int hh = (mg0 >> 5) & 31;
  const int w0 = mg0 & 31;

  const int wrow0 = (br * kG + g) * kOC + ntile * NTILE;   // weight tile rows
#if !USE_TDM
  const bf16_t* brow[2];
#pragma unroll
  for (int ns = 0; ns < 2; ++ns)
    brow[ns] = wq + (size_t)(wrow0 + ns * 16 + (lane & 15)) * kKT + (lane & 16);
#endif

  v8f acc[4][2] = {};

  auto stageA_load = [&](int k0, uint4* areg) {
    const int kpos = k0 >> 8;
    const int r = kpos / 3, s = kpos - 3 * r;
    const size_t base = ((size_t)bb * kIC) * (kHP * kWP)
                      + (size_t)(hh + r) * kWP + (w0 + s);
    const int ic0 = (k0 & 255) + kk2 * 2;
#pragma unroll
    for (int jj = 0; jj < 2; ++jj)
      areg[jj] = *(const uint4*)(xpad + base + (size_t)(ic0 + jj) * (kHP * kWP));
  };
  auto stageA_store = [&](int buf, const uint4* areg) {
#pragma unroll
    for (int jj = 0; jj < 2; ++jj) {
      Pack8 u; u.q = areg[jj];
      const int kk = kk2 * 2 + jj;
#pragma unroll
      for (int i = 0; i < 8; ++i)
        smA[buf][mblk * 8 + i][kk] = u.u[i];
    }
  };

#if USE_TDM
  auto issue_tdm = [&](int buf, int k0) {
    // D# per CDNA5 ISA ch.8: 2-D tile 64(k) x 32(oc), 2-byte elems,
    // LDS pad 4 dwords every 32 dwords -> 144 B rows (conflict-free b128 reads)
    unsigned lds_off = (unsigned)(size_t)(&smW[buf][wave][0]);
    unsigned long long ga =
        (unsigned long long)(size_t)(wq + (size_t)wrow0 * kKT + k0);
    u32x4 g0;
    g0[0] = 1u;                                        // count=1, user mode
    g0[1] = lds_off;                                   // lds_addr
    g0[2] = (unsigned)(ga & 0xFFFFFFFFu);              // global_addr lo
    g0[3] = (unsigned)((ga >> 32) & 0x1FFFFFFu) | (2u << 30); // hi + type=2
    i32x8 g1;
    g1[0] = (1 << 16)                                  // data_size = 2 B
          | (1 << 20)                                  // pad_enable
          | (4 << 22)                                  // pad_interval: 32 dw
          | (3 << 25);                                 // pad_amount: 4 dw
    g1[1] = (kKT & 0xFFFF) << 16;                      // tensor_dim0 lo16
    g1[2] = (kKT >> 16) | (32 << 16);                  // td0 hi16 | tensor_dim1
    g1[3] = (KCHUNK << 16);                            // td1 hi16=0 | tile_dim0
    g1[4] = 32;                                        // tile_dim1 | tile_dim2=0
    g1[5] = kKT;                                       // tensor_dim0_stride lo32
    g1[6] = 0;
    g1[7] = 0;
    i32x4 z4 = {0, 0, 0, 0};
    i32x8 z8 = {0, 0, 0, 0, 0, 0, 0, 0};
    __builtin_amdgcn_tensor_load_to_lds(g0, g1, z4, z4, z8, 0);
  };
#endif

  auto compute_half = [&](int buf, int h, int k0) {
    v16bf afrag[4];
#pragma unroll
    for (int ms = 0; ms < 4; ++ms) {
      const u16_t* rowp =
          &smA[buf][ms * 16 + (lane & 15)][h * 32 + ((lane & 16) ? 16 : 0)];
      ABFrag u;
      u.q[0] = *(const uint4*)(rowp);       // K 0..7   (+16 hi half-wave)
      u.q[1] = *(const uint4*)(rowp + 8);   // K 8..15  (+16 hi half-wave)
      afrag[ms] = u.v;
    }
    v16bf bfrag[2];
#pragma unroll
    for (int ns = 0; ns < 2; ++ns) {
      ABFrag u;
#if USE_TDM
      const unsigned char* wp = &smW[buf][wave]
          [(ns * 16 + (lane & 15)) * WROWB + (h * 32 + (lane & 16)) * 2];
      u.q[0] = *(const uint4*)(wp);
      u.q[1] = *(const uint4*)(wp + 16);
#else
      const bf16_t* p = brow[ns] + k0 + h * 32;
      u.q[0] = *(const uint4*)(p);
      u.q[1] = *(const uint4*)(p + 8);
#endif
      bfrag[ns] = u.v;
    }
#pragma unroll
    for (int ms = 0; ms < 4; ++ms)
#pragma unroll
      for (int ns = 0; ns < 2; ++ns)
        acc[ms][ns] = __builtin_amdgcn_wmma_f32_16x16x32_bf16(
            false, afrag[ms], false, bfrag[ns], (short)0, acc[ms][ns],
            false, false);
  };

  // ---- software-pipelined main loop: 36 chunks of K=64 (2 WMMA K-steps) ----
  uint4 areg[2];
  stageA_load(0, areg);
  stageA_store(0, areg);
#if USE_TDM
  issue_tdm(0, 0);
#endif

  for (int c = 0; c < NCHUNK; ++c) {
    const int buf = c & 1;
#if USE_TDM
    __builtin_amdgcn_s_wait_tensorcnt(0);   // own wave's W-tile(c) landed
#endif
    __syncthreads();                        // A-tile(c) staged block-wide
    const bool more = (c + 1) < NCHUNK;
    if (more) {
      stageA_load((c + 1) * KCHUNK, areg);  // prefetch next A into regs
#if USE_TDM
      issue_tdm(buf ^ 1, (c + 1) * KCHUNK); // TDM next W tile (other buffer)
#endif
    }
    compute_half(buf, 0, c * KCHUNK);
    compute_half(buf, 1, c * KCHUNK);
    if (more) stageA_store(buf ^ 1, areg);
  }

  // ---- epilogue: weight scale, psum fake-quant, group/branch reduction ----
#pragma unroll
  for (int ms = 0; ms < 4; ++ms)
#pragma unroll
    for (int ns = 0; ns < 2; ++ns)
#pragma unroll
      for (int e = 0; e < 8; ++e) {
        int ml = ms * 16 + e + ((lane & 16) ? 8 : 0);
        int nl = ns * 16 + (lane & 15);
        float t = acc[ms][ns][e] * sw;
        float q = rintf(fminf(fmaxf(t * inv_sps, -128.f), 127.f)) * sps;
        atomicAdd(&smO[ml][nl], sgn * q);   // ds_add_f32 across 8 waves
      }
  __syncthreads();

  for (int i = tid; i < MTILE * NTILE; i += THREADS) {
    int ml = i >> 5, nl = i & 31;
    int mg2 = mtile * MTILE + ml;
    int b2 = mg2 >> 10, h2 = (mg2 >> 5) & 31, w2 = mg2 & 31;
    int oc = ntile * NTILE + nl;
    out[(((size_t)b2 * kOC + oc) * kH + h2) * kW + w2] = smO[ml][nl];
  }
}

// ---------------------------------------------------------------------------
extern "C" void kernel_launch(void* const* d_in, const int* in_sizes, int n_in,
                              void* d_out, int out_size, void* d_ws, size_t ws_size,
                              hipStream_t stream) {
  const float* x      = (const float*)d_in[0];
  const float* weight = (const float*)d_in[1];
  const float* s_w_p  = (const float*)d_in[2];
  const float* s_w_n  = (const float*)d_in[3];
  const float* s_ps_p = (const float*)d_in[4];
  const float* s_ps_n = (const float*)d_in[5];
  float* out = (float*)d_out;

  bf16_t* wq   = (bf16_t*)d_ws;                         // 9.44 MB
  bf16_t* xpad = (bf16_t*)((char*)d_ws + WQ_ELEMS * 2); // 18.94 MB

  {
    long long total = (long long)WQ_ELEMS;
    int blocks = (int)((total + 255) / 256);
    quant_weights_kernel<<<blocks, 256, 0, stream>>>(weight, s_w_p, s_w_n, wq);
  }
  {
    long long total = (long long)XP_ELEMS;
    int blocks = (int)((total + 255) / 256);
    pad_cvt_kernel<<<blocks, 256, 0, stream>>>(x, xpad);
  }
  {
    dim3 grid(kB * kH * kW / MTILE, kOC / NTILE);       // (512, 8)
    conv_wmma_kernel<<<grid, THREADS, 0, stream>>>(xpad, wq, s_w_p, s_w_n,
                                                   s_ps_p, s_ps_n, out);
  }
}
